// LSTM_47064251629725
// MI455X (gfx1250) — matrix operand order repaired and checked by reading, not compiled, not run on hardware
//
#include <hip/hip_runtime.h>
#include <stdint.h>

typedef __attribute__((ext_vector_type(16))) __bf16 v16bf;
typedef __attribute__((ext_vector_type(8)))  float  v8f;
typedef __attribute__((ext_vector_type(4)))  int    v4i_;

#define T_STEPS 256
#define BATCH   64
#define HID     1024
#define KTOT    2048
#define NWG     64              // persistent workgroups; each owns 16 h-columns
#define COLS_PER_WG 64          // 4 gates x 16 columns
#define COL_STRIDE_B (KTOT*2 + 16)   // bytes per panel column in LDS (16B pad)
#define PANEL_BYTES (COLS_PER_WG * COL_STRIDE_B)          // 263,168
#define BIAS_OFF    PANEL_BYTES
#define G_OFF       (PANEL_BYTES + 256)
#define G_STRIDE    72          // floats, padded row stride of gates scratch
#define SMEM_BYTES  (G_OFF + BATCH * G_STRIDE * 4)        // 281,856 < 320K

__device__ __forceinline__ unsigned short f2bf(float f) {
  unsigned int u = __float_as_uint(f);
  unsigned int r = u + 0x7FFFu + ((u >> 16) & 1u);   // round-to-nearest-even
  return (unsigned short)(r >> 16);
}

__device__ __forceinline__ float fast_sigmoid(float x) {
  return __fdividef(1.f, 1.f + __expf(-x));
}
__device__ __forceinline__ float fast_tanh(float x) {
  // tanh(x) = 1 - 2/(exp(2x)+1); saturates correctly for |x| large
  return 1.f - __fdividef(2.f, 1.f + __expf(2.f * x));
}

// A-frag: lane holds K blocks {base..base+7} and {base+16..base+23} (16 bf16)
__device__ __forceinline__ v16bf load_a_frag(const unsigned short* p) {
  union { uint4 u[2]; v16bf v; } r;
  r.u[0] = *reinterpret_cast<const uint4*>(p);
  r.u[1] = *reinterpret_cast<const uint4*>(p + 16);
  return r.v;
}
// B-frag: lane holds K blocks {base..base+7} and {base+16..base+23} for its column
__device__ __forceinline__ v16bf load_b_frag(const void* p) {
  union { uint4 u[2]; v16bf v; } r;
  r.u[0] = *reinterpret_cast<const uint4*>(p);
  r.u[1] = *reinterpret_cast<const uint4*>(reinterpret_cast<const char*>(p) + 32);
  return r.v;
}

// ---------------- prep kernels ----------------
__global__ void conv_bf16(const float* __restrict__ src,
                          unsigned short* __restrict__ dst, int n) {
  int i = blockIdx.x * blockDim.x + threadIdx.x;
  int stride = gridDim.x * blockDim.x;
  for (; i < n; i += stride) dst[i] = f2bf(src[i]);
}

// Pack W so WG wg gets 64 contiguous columns: col = gate*16 + jj,
// source row = gate_base + wg*16 + jj, K contiguous. Gate order f,i,o,g.
__global__ void pack_w(const float* __restrict__ Wf, const float* __restrict__ Wi,
                       const float* __restrict__ Wo, const float* __restrict__ Wc,
                       unsigned short* __restrict__ Wpack) {
  const int total = NWG * COLS_PER_WG * KTOT;   // 8,388,608
  int i = blockIdx.x * blockDim.x + threadIdx.x;
  int stride = gridDim.x * blockDim.x;
  for (; i < total; i += stride) {
    int k   = i & (KTOT - 1);
    int col = (i >> 11) & 63;
    int wg  = i >> 17;
    int g   = col >> 4, jj = col & 15;
    int row = wg * 16 + jj;
    const float* src = (g == 0) ? Wf : (g == 1) ? Wi : (g == 2) ? Wo : Wc;
    Wpack[i] = f2bf(src[(size_t)row * KTOT + k]);
  }
}

// ---------------- persistent LSTM kernel ----------------
__global__ void __launch_bounds__(256, 1)
lstm_persistent(const unsigned short* __restrict__ xbf,
                const unsigned short* __restrict__ Wpack,
                unsigned short* __restrict__ hbf,       // 2 buffers of 64*1024
                const float* __restrict__ bfv, const float* __restrict__ biv,
                const float* __restrict__ bov, const float* __restrict__ bcv,
                float* __restrict__ cOut, float* __restrict__ hOut,
                unsigned int* __restrict__ syncCnt) {
  extern __shared__ char smem[];
  float* bias = (float*)(smem + BIAS_OFF);
  float* G    = (float*)(smem + G_OFF);

  const int tid  = threadIdx.x;
  const int wg   = blockIdx.x;
  const int lane = tid & 31;
  const int wave = tid >> 5;
  // wave tile: M32 x N16 — one B-frag feeds two WMMAs (halves LDS traffic)
  const int mbase = (wave & 1) << 5;    // 0 or 32
  const int nbase = (wave >> 1) << 4;   // 0,16,32,48

  // Stage this WG's 256KB weight panel into LDS once (reused all 256 steps),
  // using the CDNA5 async global->LDS copy path (ASYNCcnt-tracked).
  {
    const unsigned short* src = Wpack + (size_t)wg * COLS_PER_WG * KTOT;
    for (int idx = tid; idx < COLS_PER_WG * (KTOT / 8); idx += 256) {
      int col = idx >> 8, chunk = idx & 255;
      char* lds = smem + col * COL_STRIDE_B + chunk * 16;
      const unsigned short* gsrc = src + idx * 8;
#if __has_builtin(__builtin_amdgcn_global_load_async_to_lds_b128)
      __builtin_amdgcn_global_load_async_to_lds_b128(
          (__attribute__((address_space(1))) v4i_*)(gsrc),
          (__attribute__((address_space(3))) v4i_*)(lds), 0, 0);
#else
      *(uint4*)lds = *(const uint4*)gsrc;
#endif
    }
#if __has_builtin(__builtin_amdgcn_global_load_async_to_lds_b128)
    asm volatile("s_wait_asynccnt 0" ::: "memory");
#endif
  }
  if (tid < 64) {
    int g = tid >> 4, jj = tid & 15;
    int row = wg * 16 + jj;
    const float* bp = (g == 0) ? bfv : (g == 1) ? biv : (g == 2) ? bov : bcv;
    bias[tid] = bp[row];
  }
  __syncthreads();

  const int khsel = lane >> 4;        // 0/1 selects K-half per ISA layout
  const int lcol  = lane & 15;
  const int arow0 = mbase + lcol;          // A rows for acc0
  const int arow1 = mbase + 16 + lcol;     // A rows for acc1
  const int aoff  = khsel * 8;
  const char* bcol = smem + (nbase + lcol) * COL_STRIDE_B + khsel * 16;
  const float biasv = bias[nbase + lcol];  // same column for both accumulators

  float creg[4] = {0.f, 0.f, 0.f, 0.f};

  for (int t = 0; t < T_STEPS; ++t) {
    const unsigned short* hprev = hbf + (size_t)(t & 1) * (BATCH * HID);
    unsigned short*       hnext = hbf + (size_t)((t + 1) & 1) * (BATCH * HID);

    v8f acc0, acc1;
    #pragma unroll
    for (int v = 0; v < 8; ++v) { acc0[v] = biasv; acc1[v] = biasv; }

    // K = 0..1023 : x_t contribution (A from global/L2, B from LDS panel)
    const unsigned short* ab0 =
        xbf + (size_t)arow0 * (T_STEPS * HID) + (size_t)t * HID + aoff;
    const unsigned short* ab1 =
        xbf + (size_t)arow1 * (T_STEPS * HID) + (size_t)t * HID + aoff;
    #pragma unroll 2
    for (int k0 = 0; k0 < 1024; k0 += 32) {
      v16bf a0 = load_a_frag(ab0 + k0);
      v16bf a1 = load_a_frag(ab1 + k0);
      v16bf b  = load_b_frag(bcol + k0 * 2);
      acc0 = __builtin_amdgcn_wmma_f32_16x16x32_bf16(false, a0, false, b,
                                                     (short)0, acc0, false, false);
      acc1 = __builtin_amdgcn_wmma_f32_16x16x32_bf16(false, a1, false, b,
                                                     (short)0, acc1, false, false);
    }
    // K = 1024..2047 : h_{t-1} contribution
    const unsigned short* hb0 = hprev + (size_t)arow0 * HID + aoff;
    const unsigned short* hb1 = hprev + (size_t)arow1 * HID + aoff;
    #pragma unroll 2
    for (int k0 = 0; k0 < 1024; k0 += 32) {
      v16bf a0 = load_a_frag(hb0 + k0);
      v16bf a1 = load_a_frag(hb1 + k0);
      v16bf b  = load_b_frag(bcol + (k0 + 1024) * 2);
      acc0 = __builtin_amdgcn_wmma_f32_16x16x32_bf16(false, a0, false, b,
                                                     (short)0, acc0, false, false);
      acc1 = __builtin_amdgcn_wmma_f32_16x16x32_bf16(false, a1, false, b,
                                                     (short)0, acc1, false, false);
    }

    // Prefetch next timestep's x rows into cache while we do pointwise+barrier
    if (t + 1 < T_STEPS) {
      __builtin_prefetch(ab0 + HID, 0, 1);
      __builtin_prefetch(ab1 + HID, 0, 1);
    }

    // Scatter gates to LDS (C layout: VGPR v -> M = v + 8*khsel, lane -> N)
    #pragma unroll
    for (int v = 0; v < 8; ++v) {
      G[(mbase + v + khsel * 8)      * G_STRIDE + nbase + lcol] = acc0[v];
      G[(mbase + 16 + v + khsel * 8) * G_STRIDE + nbase + lcol] = acc1[v];
    }
    __syncthreads();

    // Pointwise gate update: 4 (m,jj) pairs per thread; c lives in registers
    const bool last = (t == T_STEPS - 1);
    #pragma unroll
    for (int q = 0; q < 4; ++q) {
      int p = tid * 4 + q;
      int m = p >> 4, jj = p & 15;
      float f = G[m * G_STRIDE + jj];
      float i = G[m * G_STRIDE + 16 + jj];
      float o = G[m * G_STRIDE + 32 + jj];
      float g = G[m * G_STRIDE + 48 + jj];
      f = fast_sigmoid(f);
      i = fast_sigmoid(i);
      o = fast_sigmoid(o);
      g = fast_tanh(g);
      float c = f * creg[q] + i * g;
      creg[q] = c;
      float h = o * fast_tanh(c);
      int jcol = wg * 16 + jj;
      hnext[m * HID + jcol] = f2bf(h);
      if (last) { cOut[m * HID + jcol] = c; hOut[m * HID + jcol] = h; }
    }

    // Grid-wide barrier (monotonic counter; all 64 WGs are resident)
    __threadfence();   // release: publish hnext to device scope
    __syncthreads();
    if (tid == 0) {
      __hip_atomic_fetch_add(syncCnt, 1u, __ATOMIC_ACQ_REL, __HIP_MEMORY_SCOPE_AGENT);
      unsigned target = (unsigned)NWG * (unsigned)(t + 1);
      while (__hip_atomic_load(syncCnt, __ATOMIC_ACQUIRE, __HIP_MEMORY_SCOPE_AGENT) < target)
        __builtin_amdgcn_s_sleep(2);
    }
    __syncthreads();
    __threadfence();   // acquire: invalidate so all lanes see other WGPs' h
  }
}

// ---------------- final projection: last_out = h @ V^T + Vb ----------------
__global__ void __launch_bounds__(256)
out_gemm(const unsigned short* __restrict__ hbf,   // final h (buffer 0), bf16
         const unsigned short* __restrict__ Vbf,   // V_w bf16, row-major (N,K)
         const float* __restrict__ Vb,
         float* __restrict__ out) {
  const int tid = threadIdx.x, lane = tid & 31, wave = tid >> 5;
  const int mbase = (wave & 1) << 5;
  const int khsel = lane >> 4, lcol = lane & 15;
  const int arow0 = mbase + lcol, arow1 = mbase + 16 + lcol;
  const int n0 = blockIdx.x * 64 + (wave >> 1) * 16 + lcol;

  v8f acc0, acc1;
  const float bv = Vb[n0];
  #pragma unroll
  for (int v = 0; v < 8; ++v) { acc0[v] = bv; acc1[v] = bv; }

  const unsigned short* ab0 = hbf + (size_t)arow0 * HID + khsel * 8;
  const unsigned short* ab1 = hbf + (size_t)arow1 * HID + khsel * 8;
  const unsigned short* bb  = Vbf + (size_t)n0 * HID + khsel * 8;
  #pragma unroll 2
  for (int k0 = 0; k0 < 1024; k0 += 32) {
    v16bf a0 = load_a_frag(ab0 + k0);
    v16bf a1 = load_a_frag(ab1 + k0);
    v16bf b  = load_b_frag(bb + k0);
    acc0 = __builtin_amdgcn_wmma_f32_16x16x32_bf16(false, a0, false, b,
                                                   (short)0, acc0, false, false);
    acc1 = __builtin_amdgcn_wmma_f32_16x16x32_bf16(false, a1, false, b,
                                                   (short)0, acc1, false, false);
  }
  #pragma unroll
  for (int v = 0; v < 8; ++v) {
    out[(size_t)(mbase + v + khsel * 8)      * HID + n0] = acc0[v];
    out[(size_t)(mbase + 16 + v + khsel * 8) * HID + n0] = acc1[v];
  }
}

// ---------------- host launcher ----------------
extern "C" void kernel_launch(void* const* d_in, const int* in_sizes, int n_in,
                              void* d_out, int out_size, void* d_ws, size_t ws_size,
                              hipStream_t stream) {
  const float* x  = (const float*)d_in[0];
  const float* Wf = (const float*)d_in[1];
  const float* bf = (const float*)d_in[2];
  const float* Wi = (const float*)d_in[3];
  const float* bi = (const float*)d_in[4];
  const float* Wc = (const float*)d_in[5];
  const float* bc = (const float*)d_in[6];
  const float* Wo = (const float*)d_in[7];
  const float* bo = (const float*)d_in[8];
  const float* Vw = (const float*)d_in[9];
  const float* Vb = (const float*)d_in[10];

  char* ws = (char*)d_ws;
  unsigned short* xbf   = (unsigned short*)(ws);              // 33,554,432 B
  unsigned short* Wpack = (unsigned short*)(ws + 33554432);   // 16,777,216 B
  unsigned short* Vbf   = (unsigned short*)(ws + 50331648);   //  2,097,152 B
  unsigned short* hbf   = (unsigned short*)(ws + 52428800);   //    262,144 B (2 bufs)
  unsigned int*   syncC = (unsigned int*)  (ws + 52690944);   //        256 B

  (void)hipMemsetAsync(hbf, 0, 2 * BATCH * HID * sizeof(unsigned short), stream);
  (void)hipMemsetAsync(syncC, 0, 256, stream);

  conv_bf16<<<2048, 256, 0, stream>>>(x, xbf, BATCH * T_STEPS * HID);
  pack_w<<<2048, 256, 0, stream>>>(Wf, Wi, Wo, Wc, Wpack);
  conv_bf16<<<1024, 256, 0, stream>>>(Vw, Vbf, HID * HID);

  (void)hipFuncSetAttribute(reinterpret_cast<const void*>(lstm_persistent),
                            hipFuncAttributeMaxDynamicSharedMemorySize,
                            (int)SMEM_BYTES);

  float* outF = (float*)d_out;
  lstm_persistent<<<NWG, 256, SMEM_BYTES, stream>>>(
      xbf, Wpack, hbf, bf, bi, bo, bc,
      outF /*c*/, outF + BATCH * HID /*h*/, syncC);

  // T=256 is even -> final h lives in buffer 0
  out_gemm<<<16, 256, 0, stream>>>(hbf, Vbf, Vb, outF + 2 * BATCH * HID);
}
